// TimeOrientedGAT_57131654972190
// MI455X (gfx1250) — matrix operand order repaired and checked by reading, not compiled, not run on hardware
//
#include <hip/hip_runtime.h>
#include <math.h>

// ---------------------------------------------------------------------------
// CDNA5 (gfx1250) TimeOrientedGAT — all GEMMs via v_wmma_f32_16x16x32_bf16.
// LDS tiles are stored in per-lane FRAGMENT ORDER so each wmma operand is
// fetched with two ds_load_b128 per lane (no u16 gather / v_mov_b16 packing).
// ---------------------------------------------------------------------------

typedef __attribute__((ext_vector_type(16))) __bf16 v16bf;
typedef __attribute__((ext_vector_type(2)))  __bf16 v2bf;
typedef __attribute__((ext_vector_type(8)))  float  v8f;

#define B_  16
#define T_  128
#define D_  256
#define H_  64
#define C1_ 64
#define C2_ 20
#define C2P 32        // C2 padded to 32 for WMMA N-tiling
#define F1_ (H_*C1_)  // 4096
#define F2_ (H_*C2_)  // 1280
#define OW_ 32

// ---- fragment-order LDS indexing (ISA 7.12.2, wave32) ----------------------
// A 16x32 tile, element (row,k): lane = row + ((k>>3)&1)*16, elem = ((k>>4)<<3)+(k&7)
__device__ __forceinline__ int a_idx(int row, int k) {
  const int lane = (row & 15) + (((k >> 3) & 1) << 4);
  const int elem = ((k >> 4) << 3) + (k & 7);
  return (lane << 4) + elem;
}
// B 32x16 tile, element (k,col): lane = col + ((k>>3)&1)*16, elem = ((k>>4)<<3)+(k&7)
__device__ __forceinline__ int b_idx(int k, int col) {
  const int lane = (col & 15) + (((k >> 3) & 1) << 4);
  const int elem = ((k >> 4) << 3) + (k & 7);
  return (lane << 4) + elem;
}

__device__ __forceinline__ v8f wmma_bf16(v16bf a, v16bf b, v8f c) {
  // (neg_a, A, neg_b, B, c_mod, C, reuse_a, reuse_b)
  return __builtin_amdgcn_wmma_f32_16x16x32_bf16(false, a, false, b, (short)0, c,
                                                 false, false);
}

__device__ __forceinline__ v2bf pack2f(float a, float b) {
  v2bf p; p[0] = (__bf16)a; p[1] = (__bf16)b; return p;
}

// ---------------------------------------------------------------------------
// K1: Wh1[b,h] = x[b](128x256) @ W1[h](256x64) + b1[h]   -> bf16 [B,H,T,C1]
// ---------------------------------------------------------------------------
__global__ void __launch_bounds__(256)
k_wh1(const float* __restrict__ x, const float* __restrict__ W1,
      const float* __restrict__ b1, __bf16* __restrict__ wh1) {
  __shared__ v16bf As[8 * 32];   // 8 m-tiles x 32 lanes (one K-chunk)
  __shared__ v16bf Bs[4 * 32];   // 4 n-tiles
  const int bh = blockIdx.x, b = bh / H_, h = bh % H_;
  const int tid = threadIdx.x, lane = tid & 31, w = tid >> 5;
  const float* xb = x + (size_t)b * T_ * D_;
  const float* wm = W1 + (size_t)h * D_ * C1_;
  __bf16* a8 = (__bf16*)As;
  __bf16* b8 = (__bf16*)Bs;
  v8f acc[4] = {};
  for (int kc = 0; kc < D_; kc += 32) {
    for (int i = tid; i < (T_ * 32) / 2; i += 256) {       // A stage (pairs)
      const int r = i >> 4, k = (i & 15) << 1;
      *(v2bf*)(a8 + (r >> 4) * 512 + a_idx(r & 15, k)) =
          pack2f(xb[(size_t)r * D_ + kc + k], xb[(size_t)r * D_ + kc + k + 1]);
    }
    for (int i = tid; i < (32 * C1_) / 2; i += 256) {      // B stage (pairs along k)
      const int n = i % C1_, k = (i / C1_) << 1;
      *(v2bf*)(b8 + (n >> 4) * 512 + b_idx(k, n & 15)) =
          pack2f(wm[(size_t)(kc + k) * C1_ + n], wm[(size_t)(kc + k + 1) * C1_ + n]);
    }
    __syncthreads();
    const v16bf a = As[w * 32 + lane];                     // 2x ds_load_b128
#pragma unroll
    for (int nb = 0; nb < 4; ++nb)
      acc[nb] = wmma_bf16(a, Bs[nb * 32 + lane], acc[nb]);
    __syncthreads();
  }
  const int half = lane >> 4, col = lane & 15;
  __bf16* out = wh1 + (size_t)bh * T_ * C1_;
#pragma unroll
  for (int nb = 0; nb < 4; ++nb) {
    const int n = nb * 16 + col;
    const float bias = b1[h * C1_ + n];
#pragma unroll
    for (int v = 0; v < 8; ++v) {
      const int m = w * 16 + half * 8 + v;
      out[(size_t)m * C1_ + n] = (__bf16)(acc[nb][v] + bias);
    }
  }
}

// ---------------------------------------------------------------------------
// K2/K5: src/dst scores: s[bht] = Wh[bht,:] . a_src[h,:]   (VALU, packed reads)
// ---------------------------------------------------------------------------
__global__ void __launch_bounds__(256)
k_scores(const __bf16* __restrict__ wh, const float* __restrict__ as_,
         const float* __restrict__ ad_, float* __restrict__ s,
         float* __restrict__ d, int C, int ldw) {
  const int idx = blockIdx.x * 256 + threadIdx.x;  // (b*H + h)*T + t
  const int h = (idx / T_) % H_;
  const __bf16* row = wh + (size_t)idx * ldw;
  float sv = 0.f, dv = 0.f;
  for (int c = 0; c < C; c += 2) {
    const v2bf p = *(const v2bf*)(row + c);
    const float v0 = (float)p[0], v1 = (float)p[1];
    sv += v0 * as_[h * C + c] + v1 * as_[h * C + c + 1];
    dv += v0 * ad_[h * C + c] + v1 * ad_[h * C + c + 1];
  }
  s[idx] = sv;
  d[idx] = dv;
}

// ---------------------------------------------------------------------------
// K3/K6: fused leaky-relu + softmax + (P @ Wh) + relu + sigmoid per (b,h).
// P is built UNNORMALIZED directly in A-fragment layout in LDS; the softmax
// row normalization is folded into the WMMA epilogue (inv_sum per output row).
// ---------------------------------------------------------------------------
template <int CN, int CV, int FOUT>
__global__ void __launch_bounds__(256)
k_attn(const float* __restrict__ sg, const float* __restrict__ dg,
       const float* __restrict__ ab, const __bf16* __restrict__ whg,
       __bf16* __restrict__ cout) {
  constexpr int NB = CN / 16;
  __shared__ v16bf P[8 * 4 * 32];      // [mb][kb][lane]  (32 KB)
  __shared__ v16bf Ws[4 * NB * 32];    // [kb][nb][lane]
  __shared__ float ss[T_], ds[T_], inv[T_];
  const int bh = blockIdx.x, b = bh / H_, h = bh % H_;
  const int tid = threadIdx.x, lane = tid & 31, w = tid >> 5;
  if (tid < T_) { ss[tid] = sg[bh * T_ + tid]; ds[tid] = dg[bh * T_ + tid]; }
  __syncthreads();
  const float abv = ab[h];
  __bf16* p8 = (__bf16*)P;
  if (tid < T_) {                      // one attention row per thread
    const int r = tid;
    const float sr = ss[r] + abv;
    float mx = -1e30f;
    for (int j = 0; j < T_; ++j) {
      float e = sr + ds[j];
      e = (e >= 0.f) ? e : 0.2f * e;   // leaky_relu
      mx = fmaxf(mx, e);
    }
    float sum = 0.f;
    for (int j = 0; j < T_; j += 2) {
      float e0 = sr + ds[j],     e1 = sr + ds[j + 1];
      e0 = (e0 >= 0.f) ? e0 : 0.2f * e0;
      e1 = (e1 >= 0.f) ? e1 : 0.2f * e1;
      const float p0 = __expf(e0 - mx), p1 = __expf(e1 - mx);
      sum += p0 + p1;
      *(v2bf*)(p8 + ((r >> 4) * 4 + (j >> 5)) * 512 + a_idx(r & 15, j & 31)) =
          pack2f(p0, p1);
    }
    inv[r] = 1.f / sum;
  }
  __bf16* w8 = (__bf16*)Ws;
  for (int i = tid; i < (T_ * CN) / 2; i += 256) {   // stage Wh (pairs along j)
    const int c = i % CN, j = (i / CN) << 1;
    const __bf16* src = whg + ((size_t)bh * T_ + j) * CN + c;
    v2bf p; p[0] = src[0]; p[1] = src[CN];
    *(v2bf*)(w8 + ((j >> 5) * NB + (c >> 4)) * 512 + b_idx(j & 31, c & 15)) = p;
  }
  __syncthreads();
  v8f acc[NB] = {};
#pragma unroll
  for (int kb = 0; kb < 4; ++kb) {
    const v16bf a = P[(w * 4 + kb) * 32 + lane];
#pragma unroll
    for (int nb = 0; nb < NB; ++nb)
      acc[nb] = wmma_bf16(a, Ws[(kb * NB + nb) * 32 + lane], acc[nb]);
  }
  const int half = lane >> 4, col = lane & 15;
#pragma unroll
  for (int nb = 0; nb < NB; ++nb) {
    const int n = nb * 16 + col;
    if (n < CV) {
#pragma unroll
      for (int v = 0; v < 8; ++v) {
        const int m = w * 16 + half * 8 + v;
        const float xv = fmaxf(acc[nb][v] * inv[m], 0.f);   // normalize + relu
        const float sgm = 1.f / (1.f + __expf(-xv));        // sigmoid
        cout[((size_t)b * T_ + m) * FOUT + h * CV + n] = (__bf16)sgm;
      }
    }
  }
}

// ---------------------------------------------------------------------------
// K4: Wh2[b,h] = c1[b](128x4096) @ W2[h](4096x20->pad32) + b2[h]  (dominant)
// ---------------------------------------------------------------------------
__global__ void __launch_bounds__(256)
k_wh2(const __bf16* __restrict__ c1, const float* __restrict__ W2,
      const float* __restrict__ b2, __bf16* __restrict__ wh2) {
  __shared__ v16bf As[8 * 32];
  __shared__ v16bf Bs[2 * 32];
  const int bh = blockIdx.x, b = bh / H_, h = bh % H_;
  const int tid = threadIdx.x, lane = tid & 31, w = tid >> 5;
  const __bf16* cb = c1 + (size_t)b * T_ * F1_;
  const float* w2 = W2 + (size_t)h * F1_ * C2_;
  __bf16* a8 = (__bf16*)As;
  __bf16* b8 = (__bf16*)Bs;
  v8f acc[2] = {};
  for (int kc = 0; kc < F1_; kc += 32) {
    for (int i = tid; i < (T_ * 32) / 2; i += 256) {   // A: bf16 source, b32 loads
      const int r = i >> 4, k = (i & 15) << 1;
      *(v2bf*)(a8 + (r >> 4) * 512 + a_idx(r & 15, k)) =
          *(const v2bf*)(cb + (size_t)r * F1_ + kc + k);
    }
    for (int i = tid; i < (32 * C2P) / 2; i += 256) {  // B: fp32, zero-pad n>=20
      const int n = i % C2P, k = (i / C2P) << 1;
      v2bf p;
      p[0] = (n < C2_) ? (__bf16)w2[(size_t)(kc + k) * C2_ + n] : (__bf16)0.0f;
      p[1] = (n < C2_) ? (__bf16)w2[(size_t)(kc + k + 1) * C2_ + n] : (__bf16)0.0f;
      *(v2bf*)(b8 + (n >> 4) * 512 + b_idx(k, n & 15)) = p;
    }
    if (kc + 32 < F1_)  // hint next A chunk (global_prefetch_b8)
      __builtin_prefetch(cb + (size_t)(tid >> 1) * F1_ + kc + 32, 0, 0);
    __syncthreads();
    const v16bf a = As[w * 32 + lane];
#pragma unroll
    for (int nb = 0; nb < 2; ++nb)
      acc[nb] = wmma_bf16(a, Bs[nb * 32 + lane], acc[nb]);
    __syncthreads();
  }
  const int half = lane >> 4, col = lane & 15;
  __bf16* out = wh2 + (size_t)bh * T_ * C2P;
#pragma unroll
  for (int nb = 0; nb < 2; ++nb) {
    const int n = nb * 16 + col;
    const float bias = (n < C2_) ? b2[h * C2_ + n] : 0.f;
#pragma unroll
    for (int v = 0; v < 8; ++v) {
      const int m = w * 16 + half * 8 + v;
      const float val = (n < C2_) ? acc[nb][v] + bias : 0.f;  // keep pad zero
      out[(size_t)m * C2P + n] = (__bf16)val;
    }
  }
}

// ---------------------------------------------------------------------------
// K7: c3[b] = c2[b](128x1280) @ sl_w(1280x256) + sl_b ; 4 N-panels per b
// ---------------------------------------------------------------------------
__global__ void __launch_bounds__(256)
k_skip(const __bf16* __restrict__ c2, const float* __restrict__ slw,
       const float* __restrict__ slb, __bf16* __restrict__ c3) {
  __shared__ v16bf As[8 * 32];
  __shared__ v16bf Bs[4 * 32];
  const int b = blockIdx.x >> 2;
  const int N0 = (blockIdx.x & 3) * 64;
  const int tid = threadIdx.x, lane = tid & 31, w = tid >> 5;
  const __bf16* cb = c2 + (size_t)b * T_ * F2_;
  __bf16* a8 = (__bf16*)As;
  __bf16* b8 = (__bf16*)Bs;
  v8f acc[4] = {};
  for (int kc = 0; kc < F2_; kc += 32) {
    for (int i = tid; i < (T_ * 32) / 2; i += 256) {
      const int r = i >> 4, k = (i & 15) << 1;
      *(v2bf*)(a8 + (r >> 4) * 512 + a_idx(r & 15, k)) =
          *(const v2bf*)(cb + (size_t)r * F2_ + kc + k);
    }
    for (int i = tid; i < (32 * 64) / 2; i += 256) {
      const int n = i % 64, k = (i / 64) << 1;
      *(v2bf*)(b8 + (n >> 4) * 512 + b_idx(k, n & 15)) =
          pack2f(slw[(size_t)(kc + k) * D_ + N0 + n],
                 slw[(size_t)(kc + k + 1) * D_ + N0 + n]);
    }
    __syncthreads();
    const v16bf a = As[w * 32 + lane];
#pragma unroll
    for (int nb = 0; nb < 4; ++nb)
      acc[nb] = wmma_bf16(a, Bs[nb * 32 + lane], acc[nb]);
    __syncthreads();
  }
  const int half = lane >> 4, col = lane & 15;
#pragma unroll
  for (int nb = 0; nb < 4; ++nb) {
    const int dcol = N0 + nb * 16 + col;
    const float bias = slb[dcol];
#pragma unroll
    for (int v = 0; v < 8; ++v) {
      const int m = w * 16 + half * 8 + v;
      c3[((size_t)b * T_ + m) * D_ + dcol] = (__bf16)(acc[nb][v] + bias);
    }
  }
}

// ---------------------------------------------------------------------------
// K8: out[b] = gar_w^T(32x128) @ c3[b](128x256) + gar_b  -> fp32 [B,32,256]
// ---------------------------------------------------------------------------
__global__ void __launch_bounds__(256)
k_gar(const __bf16* __restrict__ c3, const float* __restrict__ garw,
      const float* __restrict__ garb, float* __restrict__ out) {
  __shared__ v16bf As[2 * 32];     // 2 m-tiles (o)
  __shared__ v16bf Bs[16 * 32];    // 16 n-tiles (d)
  const int b = blockIdx.x;
  const int tid = threadIdx.x, lane = tid & 31, w = tid >> 5;
  const int mb = w >> 2, nb0 = (w & 3) * 4;  // 2 M-blocks x 16 N-blocks / 8 waves
  __bf16* a8 = (__bf16*)As;
  __bf16* b8 = (__bf16*)Bs;
  v8f acc[4] = {};
  for (int kc = 0; kc < T_; kc += 32) {
    for (int i = tid; i < (OW_ * 32) / 2; i += 256) {   // A = gar_w^T chunk
      const int o = i >> 4, t = (i & 15) << 1;
      *(v2bf*)(a8 + (o >> 4) * 512 + a_idx(o & 15, t)) =
          pack2f(garw[(size_t)(kc + t) * OW_ + o],
                 garw[(size_t)(kc + t + 1) * OW_ + o]);
    }
    for (int i = tid; i < (32 * D_) / 2; i += 256) {    // B = c3 chunk (bf16)
      const int dc = i % D_, k = (i / D_) << 1;
      v2bf p;
      p[0] = c3[((size_t)b * T_ + kc + k) * D_ + dc];
      p[1] = c3[((size_t)b * T_ + kc + k + 1) * D_ + dc];
      *(v2bf*)(b8 + (dc >> 4) * 512 + b_idx(k, dc & 15)) = p;
    }
    __syncthreads();
    const v16bf a = As[mb * 32 + lane];
#pragma unroll
    for (int j = 0; j < 4; ++j)
      acc[j] = wmma_bf16(a, Bs[(nb0 + j) * 32 + lane], acc[j]);
    __syncthreads();
  }
  const int half = lane >> 4, col = lane & 15;
#pragma unroll
  for (int j = 0; j < 4; ++j) {
    const int dc = (nb0 + j) * 16 + col;
#pragma unroll
    for (int v = 0; v < 8; ++v) {
      const int o = mb * 16 + half * 8 + v;
      out[((size_t)b * OW_ + o) * D_ + dc] = acc[j][v] + garb[o];
    }
  }
}

// ---------------------------------------------------------------------------
extern "C" void kernel_launch(void* const* d_in, const int* in_sizes, int n_in,
                              void* d_out, int out_size, void* d_ws, size_t ws_size,
                              hipStream_t stream) {
  const float* x    = (const float*)d_in[0];
  const float* W1   = (const float*)d_in[1];
  const float* b1   = (const float*)d_in[2];
  const float* a1s  = (const float*)d_in[3];
  const float* a1d  = (const float*)d_in[4];
  const float* a1b  = (const float*)d_in[5];
  const float* W2   = (const float*)d_in[6];
  const float* b2   = (const float*)d_in[7];
  const float* a2s  = (const float*)d_in[8];
  const float* a2d  = (const float*)d_in[9];
  const float* a2b  = (const float*)d_in[10];
  const float* slw  = (const float*)d_in[11];
  const float* slb  = (const float*)d_in[12];
  const float* garw = (const float*)d_in[13];
  const float* garb = (const float*)d_in[14];
  float* out = (float*)d_out;

  // Workspace partition (regions reused as lifetimes end): ~35 MB
  char* ws = (char*)d_ws;
  __bf16* whX = (__bf16*)(ws);                                  // wh1(16MB) then wh2(8MB)
  __bf16* cX  = (__bf16*)(ws + (size_t)16 * 1024 * 1024);       // c1(16MB) then c2(5MB)
  float*  sS  = (float*)(ws + (size_t)32 * 1024 * 1024);        // src scores (512KB)
  float*  sD  = (float*)(ws + (size_t)33 * 1024 * 1024);        // dst scores (512KB)
  __bf16* c3  = (__bf16*)(ws + (size_t)34 * 1024 * 1024);       // c3 bf16 (1MB)

  const int nrows = B_ * H_ * T_;  // 131072

  // Layer 1
  k_wh1<<<B_ * H_, 256, 0, stream>>>(x, W1, b1, whX);
  k_scores<<<nrows / 256, 256, 0, stream>>>(whX, a1s, a1d, sS, sD, C1_, C1_);
  k_attn<C1_, C1_, F1_><<<B_ * H_, 256, 0, stream>>>(sS, sD, a1b, whX, cX);
  // Layer 2 (wh2 reuses wh1 region, c2 reuses c1 region)
  k_wh2<<<B_ * H_, 256, 0, stream>>>(cX, W2, b2, whX);
  k_scores<<<nrows / 256, 256, 0, stream>>>(whX, a2s, a2d, sS, sD, C2_, C2P);
  k_attn<C2P, C2_, F2_><<<B_ * H_, 256, 0, stream>>>(sS, sD, a2b, whX, cX);
  // Skip-linear + GAR
  k_skip<<<B_ * 4, 256, 0, stream>>>(cX, slw, slb, c3);
  k_gar<<<B_, 256, 0, stream>>>(c3, garw, garb, out);
}
